// NeuralCde_85031762526431
// MI455X (gfx1250) — matrix operand order repaired and checked by reading, not compiled, not run on hardware
//
#include <hip/hip_runtime.h>
#include <hip/hip_fp16.h>

#define T_LEN 64
#define C_DIM 32
#define H_DIM 64
#define MLPW  128
#define O_DIM 10
#define BT    16      // batch rows per workgroup -> 32 independent WGs
#define NTHR  512     // 16 waves -> 4 waves per SIMD32 for latency hiding

typedef __attribute__((ext_vector_type(16))) _Float16 v16h;
typedef __attribute__((ext_vector_type(8)))  float    v8f;

union V16H { v16h v; uint4 q[2]; };

static __device__ __forceinline__ v8f wmma_f16(v16h a, v16h b, v8f c) {
  return __builtin_amdgcn_wmma_f32_16x16x32_f16(false, a, false, b, (short)0, c, false, false);
}

// d_ws layout (in halves):
//   [0, 16*512)            : W1 swizzled, 16 tiles (nt*2+kt), 512 halves/tile
//   [16*512, +512*512)     : W2 swizzled, 512 tiles (nt*4+kt)
// Tile layout: lane n owns 16 contiguous halves at tile_base + n*16, holding
// B-operand element (K = (hj&7) + 8*(n>>4) + 16*(hj>>3) + kt*32, N = (n&15) + nt*16) = W[N][K].
#define W1SW_OFF 0
#define W2SW_OFF (16*512)

__global__ __launch_bounds__(256) void cde_prep(const float* __restrict__ W1,
                                                const float* __restrict__ W2,
                                                _Float16* __restrict__ ws) {
  int tid = blockIdx.x * 256 + threadIdx.x;
  const int total = (16 + 512) * 32;
  if (tid >= total) return;
  int tile = tid >> 5, n = tid & 31;
  int l = n & 15, hi = n >> 4;
  __attribute__((aligned(16))) _Float16 buf[16];
  _Float16* dst;
  if (tile < 16) {                       // W1 (128 x 64)
    int nt = tile >> 1, kt = tile & 1;
    int col = nt * 16 + l;
#pragma unroll
    for (int hj = 0; hj < 16; hj++) {
      int k = kt * 32 + (hj & 7) + 8 * hi + 16 * (hj >> 3);
      buf[hj] = (_Float16)W1[col * H_DIM + k];
    }
    dst = ws + W1SW_OFF + tile * 512 + n * 16;
  } else {                               // W2 (2048 x 128)
    int t2 = tile - 16;
    int nt = t2 >> 2, kt = t2 & 3;
    int col = nt * 16 + l;
#pragma unroll
    for (int hj = 0; hj < 16; hj++) {
      int k = kt * 32 + (hj & 7) + 8 * hi + 16 * (hj >> 3);
      buf[hj] = (_Float16)W2[col * MLPW + k];
    }
    dst = ws + W2SW_OFF + t2 * 512 + n * 16;
  }
  *(uint4*)dst     = *(const uint4*)buf;
  *(uint4*)(dst+8) = *(const uint4*)(buf + 8);
}

__global__ __launch_bounds__(NTHR) void cde_main(const float* __restrict__ coeffs,
                                                 const float* __restrict__ b1,
                                                 const float* __restrict__ b2,
                                                 const float* __restrict__ Wi,
                                                 const float* __restrict__ bi,
                                                 const float* __restrict__ Wr,
                                                 const float* __restrict__ br,
                                                 const _Float16* __restrict__ ws,
                                                 float* __restrict__ out) {
  __shared__ __attribute__((aligned(16))) float    zbuf[BT * H_DIM];  // z (f32 state)
  __shared__ __attribute__((aligned(16))) float    ksum[BT * H_DIM];  // RK4 running sum
  __shared__ __attribute__((aligned(16))) float    kbuf[BT * H_DIM];  // current stage k
  __shared__ __attribute__((aligned(16))) _Float16 zz[BT * H_DIM];    // f16 stage input
  __shared__ __attribute__((aligned(16))) _Float16 hb[BT * MLPW];     // f16 relu(h)
  __shared__ float dxb[BT * C_DIM];
  __shared__ float b1s[MLPW];
  __shared__ float b2s[C_DIM * H_DIM];

  const int tid  = threadIdx.x;
  const int w    = tid >> 5;      // wave 0..15
  const int lane = tid & 31;
  const int l    = lane & 15;
  const int hi   = lane >> 4;
  const int bg0  = blockIdx.x * BT;

  if (tid < MLPW) b1s[tid] = b1[tid];
#pragma unroll
  for (int j = 0; j < 4; j++) b2s[tid + NTHR * j] = b2[tid + NTHR * j];

  // W1 B-tiles resident in VGPRs: wave w<8 owns hidden cols [16w, 16w+16)
  V16H bw1[2];
  {
    int w8 = w & 7;
#pragma unroll
    for (int kt = 0; kt < 2; kt++) {
      const _Float16* p = ws + W1SW_OFF + (w8 * 2 + kt) * 512 + lane * 16;
      bw1[kt].q[0] = *(const uint4*)p;
      bw1[kt].q[1] = *(const uint4*)(p + 8);
    }
  }

  // z0 = coeffs[:,0,:] @ Wi.T + bi ; clear RK4 buffers; zz = f16(z0)
#pragma unroll
  for (int j = 0; j < 2; j++) {
    int e = tid + NTHR * j;
    int bl = e >> 6, h = e & 63;
    const float* crow = coeffs + (size_t)(bg0 + bl) * T_LEN * C_DIM;
    float a = bi[h];
#pragma unroll
    for (int c = 0; c < C_DIM; c++) a += crow[c] * Wi[h * C_DIM + c];
    zbuf[e] = a; ksum[e] = 0.f; kbuf[e] = 0.f; zz[e] = (_Float16)a;
  }
  __syncthreads();

  for (int t = 0; t < T_LEN - 1; ++t) {
    // dX on this unit segment (piecewise-constant derivative): 512 elems, 1/thread
    {
      int e = tid;
      int bl = e >> 5, c = e & 31;
      const float* cp = coeffs + (size_t)(bg0 + bl) * T_LEN * C_DIM + t * C_DIM + c;
      dxb[e] = cp[C_DIM] - cp[0];
    }

    for (int stage = 0; stage < 4; ++stage) {
      // ---- GEMM1: h = relu(zz @ W1^T + b1); waves 0..7, cols [16w,16w+16) ----
      if (w < 8) {               // wave-uniform branch: EXEC all-ones inside
        V16H a0, a1;
        {
          const _Float16* rp = zz + (size_t)l * H_DIM;
          int o = 8 * hi;
          a0.q[0] = *(const uint4*)(rp + o);       a0.q[1] = *(const uint4*)(rp + 16 + o);
          a1.q[0] = *(const uint4*)(rp + 32 + o);  a1.q[1] = *(const uint4*)(rp + 48 + o);
        }
        v8f hc = {};
        hc = wmma_f16(a0.v, bw1[0].v, hc);
        hc = wmma_f16(a1.v, bw1[1].v, hc);
        int col = w * 16 + l;
        float bias = b1s[col];
#pragma unroll
        for (int r = 0; r < 8; r++) {
          float v = hc[r] + bias;
          v = fmaxf(v, 0.f);
          hb[(r + 8 * hi) * MLPW + col] = (_Float16)v;
        }
      }
      __syncthreads();

      // ---- GEMM2 + b2 + tanh + einsum(dx) -> kbuf; wave w owns h in [4w,4w+4) ----
      V16H A[4];
      {
        const _Float16* rp = hb + (size_t)l * MLPW;
        int o = 8 * hi;
#pragma unroll
        for (int kt = 0; kt < 4; kt++) {
          A[kt].q[0] = *(const uint4*)(rp + kt * 32 + o);
          A[kt].q[1] = *(const uint4*)(rp + kt * 32 + 16 + o);
        }
      }
      for (int p = 0; p < 4; p++) {      // 4 hidden units per wave
        int hidx = w * 4 + p;
        float y[8];
#pragma unroll
        for (int r = 0; r < 8; r++) y[r] = 0.f;
#pragma unroll
        for (int half = 0; half < 2; ++half) {   // two 16-col halves of one h
          int ntg = hidx * 2 + half;
          const _Float16* bp = ws + W2SW_OFF + (size_t)ntg * 2048 + lane * 16;
          V16H B0, B1t, B2t, B3t;
          B0.q[0]  = *(const uint4*)(bp);         B0.q[1]  = *(const uint4*)(bp + 8);
          B1t.q[0] = *(const uint4*)(bp + 512);   B1t.q[1] = *(const uint4*)(bp + 520);
          B2t.q[0] = *(const uint4*)(bp + 1024);  B2t.q[1] = *(const uint4*)(bp + 1032);
          B3t.q[0] = *(const uint4*)(bp + 1536);  B3t.q[1] = *(const uint4*)(bp + 1544);
          v8f acc = {};
          acc = wmma_f16(A[0].v, B0.v,  acc);
          acc = wmma_f16(A[1].v, B1t.v, acc);
          acc = wmma_f16(A[2].v, B2t.v, acc);
          acc = wmma_f16(A[3].v, B3t.v, acc);
          int col = ntg * 16 + l;
          float bias = b2s[col];
          int cc = half * 16 + l;
#pragma unroll
          for (int r = 0; r < 8; r++)
            y[r] += tanhf(acc[r] + bias) * dxb[(r + 8 * hi) * C_DIM + cc];
        }
        // reduce over the 16 lanes (c-dimension) per 16-lane group
#pragma unroll
        for (int r = 0; r < 8; r++) {
          y[r] += __shfl_xor(y[r], 1, 32);
          y[r] += __shfl_xor(y[r], 2, 32);
          y[r] += __shfl_xor(y[r], 4, 32);
          y[r] += __shfl_xor(y[r], 8, 32);
        }
        if (l == 0) {
          int brow = 8 * hi;
#pragma unroll
          for (int r = 0; r < 8; r++) kbuf[(brow + r) * H_DIM + hidx] += y[r];
        }
      }
      __syncthreads();

      // ---- RK4 combine ----
      const float aN = (stage == 2) ? 1.0f : 0.5f;
      const float wS = (stage == 1 || stage == 2) ? 2.0f : 1.0f;
#pragma unroll
      for (int j = 0; j < 2; j++) {
        int e = tid + NTHR * j;
        float kv = kbuf[e];
        kbuf[e] = 0.f;
        if (stage < 3) {
          ksum[e] += wS * kv;
          float zn = zbuf[e] + aN * kv;
          zz[e] = (_Float16)zn;
        } else {
          float s  = ksum[e] + kv;
          float zn = zbuf[e] + s * (1.0f / 6.0f);
          zbuf[e] = zn; ksum[e] = 0.f;
          zz[e] = (_Float16)zn;
        }
      }
      __syncthreads();
    }
  }

  // readout: out = zT @ Wr.T + br
  if (tid < BT * O_DIM) {
    int bl = tid / O_DIM, o = tid % O_DIM;
    float a = br[o];
#pragma unroll
    for (int h = 0; h < H_DIM; h++) a += zbuf[bl * H_DIM + h] * Wr[o * H_DIM + h];
    out[(size_t)(bg0 + bl) * O_DIM + o] = a;
  }
}

extern "C" void kernel_launch(void* const* d_in, const int* in_sizes, int n_in,
                              void* d_out, int out_size, void* d_ws, size_t ws_size,
                              hipStream_t stream) {
  const float* coeffs = (const float*)d_in[0];
  const float* W1     = (const float*)d_in[1];
  const float* b1     = (const float*)d_in[2];
  const float* W2     = (const float*)d_in[3];
  const float* b2     = (const float*)d_in[4];
  const float* Wi     = (const float*)d_in[5];
  const float* bi     = (const float*)d_in[6];
  const float* Wr     = (const float*)d_in[7];
  const float* br     = (const float*)d_in[8];
  float* outp = (float*)d_out;
  _Float16* ws = (_Float16*)d_ws;

  // swizzle W1/W2 into WMMA B-operand layout (re-done every call: deterministic)
  hipLaunchKernelGGL(cde_prep, dim3(66), dim3(256), 0, stream, W1, W2, ws);
  // 32 independent workgroups (16 batch rows each), 16 waves per WG
  hipLaunchKernelGGL(cde_main, dim3(32), dim3(NTHR), 0, stream,
                     coeffs, b1, b2, Wi, bi, Wr, br, ws, outp);
  (void)in_sizes; (void)n_in; (void)out_size; (void)ws_size;
}
